// lstm_ode_model_49744311222339
// MI455X (gfx1250) — matrix operand order repaired and checked by reading, not compiled, not run on hardware
//
#include <hip/hip_runtime.h>

typedef _Float16 half_t;
typedef __attribute__((ext_vector_type(16))) _Float16 v16h;
typedef __attribute__((ext_vector_type(8)))  _Float16 v8h;
typedef __attribute__((ext_vector_type(8)))  float    v8f;

#define ACT_NONE 0
#define ACT_ELU  1
#define ACT_RELU 2

#define EPI_PLAIN   0   // act -> Cf/Ch
#define EPI_STAGE   1   // Cf = k;  stage_h = f16(z + alpha*k)
#define EPI_COMBINE 2   // zn = z + dtd6*(q1+2q2+2q3+acc); zout,zhout,(pred)

// Problem constants (match reference)
#define BB   128          // batch
#define TT   1024         // seq len
#define DD   128          // input size
#define HH   512          // hidden
#define NHH  1024         // 2*H
#define G4   2048         // 4*H
#define KCAT 640          // D + H
#define NSTEPS 5
#define NSUB   8

__device__ __forceinline__ float sigf(float x) { return 1.0f / (1.0f + __expf(-x)); }

// ---------------------------------------------------------------------------
// Generic WMMA GEMM:  C[M,N] = act(A[M,K](f16) @ W[N,K](f16)^T + bias)
// One wave computes a 16x64 tile; block = 4 waves (128 threads).
// Requires: M%16==0, N%64==0, K%32==0; rows 32B-aligned.
// Epilogue modes fuse the RK4 pointwise work (block-uniform branches).
// ---------------------------------------------------------------------------
__global__ __launch_bounds__(128)
void wmma_gemm_kernel(const half_t* __restrict__ A, int lda,
                      const half_t* __restrict__ W, int ldw,
                      const float* __restrict__ bias,
                      float* __restrict__ Cf, half_t* __restrict__ Ch,
                      int ldc, int M, int N, int K, int act, int epi,
                      const float* __restrict__ zvec, float alpha,
                      half_t* __restrict__ stage_h,
                      const float* __restrict__ q1, const float* __restrict__ q2,
                      const float* __restrict__ q3, float dtd6,
                      float* __restrict__ zout, half_t* __restrict__ zhout,
                      half_t* __restrict__ pred, int s)
{
    const int wave = threadIdx.x >> 5;
    const int lane = threadIdx.x & 31;
    const int nt64 = N >> 6;
    const int tile = blockIdx.x * 4 + wave;
    const int mtiles = M >> 4;
    if (tile >= mtiles * nt64) return;           // wave-uniform exit
    const int mt  = tile / nt64;
    const int nt  = tile % nt64;
    const int m0  = mt << 4;
    const int n0  = nt << 6;
    const int ncl = lane & 15;
    const int hi  = lane >> 4;
    const int mrow = m0 + ncl;

    v8f acc0 = {}, acc1 = {}, acc2 = {}, acc3 = {};

    for (int kk = 0; kk < K; kk += 32) {
        // A 16x32 f16 tile: lanes 0-15 rows; element s -> K=hi*16+s (contiguous)
        const half_t* ap = A + (size_t)mrow * lda + kk + hi * 16;
        v8h alo = *(const v8h*)(ap);
        v8h ahi = *(const v8h*)(ap + 8);
        v16h a;
#pragma unroll
        for (int i = 0; i < 8; ++i) { a[i] = alo[i]; a[8 + i] = ahi[i]; }

        // B 32x16 f16 tile: lane = column n, lane-half selects K half; 16 contiguous
        // f16 from weight row (W is [N,K] row-major).
        const int kb = kk + hi * 16;
        const half_t* wp = W + (size_t)(n0 + ncl) * ldw + kb;
        v16h b0 = *(const v16h*)(wp);
        v16h b1 = *(const v16h*)(wp + (size_t)16 * ldw);
        v16h b2 = *(const v16h*)(wp + (size_t)32 * ldw);
        v16h b3 = *(const v16h*)(wp + (size_t)48 * ldw);

        acc0 = __builtin_amdgcn_wmma_f32_16x16x32_f16(false, a, false, b0, (short)0, acc0, false, false);
        acc1 = __builtin_amdgcn_wmma_f32_16x16x32_f16(false, a, false, b1, (short)0, acc1, false, false);
        acc2 = __builtin_amdgcn_wmma_f32_16x16x32_f16(false, a, false, b2, (short)0, acc2, false, false);
        acc3 = __builtin_amdgcn_wmma_f32_16x16x32_f16(false, a, false, b3, (short)0, acc3, false, false);
    }

    v8f accs[4] = {acc0, acc1, acc2, acc3};
#pragma unroll
    for (int t4 = 0; t4 < 4; ++t4) {
        const int n = n0 + t4 * 16 + ncl;
        const float bv = bias ? bias[n] : 0.0f;
#pragma unroll
        for (int r = 0; r < 8; ++r) {
            float v = accs[t4][r] + bv;
            const int m = m0 + hi * 8 + r;          // C/D layout: VGPR r -> row (hi*8+r)
            const size_t idx = (size_t)m * ldc + n;
            if (epi == EPI_PLAIN) {
                if (act == ACT_ELU)  v = (v > 0.0f) ? v : (__expf(v) - 1.0f);
                if (act == ACT_RELU) v = (v > 0.0f) ? v : 0.0f;
                if (Cf) Cf[idx] = v;
                if (Ch) Ch[idx] = (half_t)v;
            } else if (epi == EPI_STAGE) {
                Cf[idx] = v;                                       // k_i
                stage_h[idx] = (half_t)(zvec[idx] + alpha * v);    // next stage input
            } else { // EPI_COMBINE (v == k4 element)
                const float zn = zvec[idx] +
                    dtd6 * (q1[idx] + 2.0f * q2[idx] + 2.0f * q3[idx] + v);
                zout[idx]  = zn;
                zhout[idx] = (half_t)zn;
                if (pred) pred[((size_t)m * NSTEPS + s) * HH + n] = (half_t)zn;
            }
        }
    }
}

// ---------------------------------------------------------------------------
// Fully fused LSTM step: gates = [x_t | h_{t-1}] @ Wcat^T + bias, then the
// gate nonlinearity + cell/hidden update, in ONE kernel.
// Each wave owns a 16(batch) x 16(j) slice and computes the four gate tiles
// at N = j, H+j, 2H+j, 3H+j, so the epilogue has i/f/g/o for its elements.
// x_t is read from fp32 `inputs` with in-register conversion (no pack pass).
// h ping-pongs between two buffers (read hprev, write hnext).
// ---------------------------------------------------------------------------
__global__ __launch_bounds__(128)
void lstm_gemm_kernel(const float* __restrict__ inputs, int t,
                      const half_t* __restrict__ hprev,
                      const half_t* __restrict__ wcat,
                      const float* __restrict__ biascat,
                      float* __restrict__ c, half_t* __restrict__ hnext)
{
    const int wave = threadIdx.x >> 5;
    const int lane = threadIdx.x & 31;
    const int tile = blockIdx.x * 4 + wave;      // 8 mtiles * 32 jtiles = 256
    if (tile >= (BB / 16) * (HH / 16)) return;
    const int mt = tile / (HH / 16);
    const int jt = tile % (HH / 16);
    const int m0 = mt << 4;
    const int j0 = jt << 4;
    const int ncl = lane & 15;
    const int hi  = lane >> 4;
    const int mrow = m0 + ncl;                   // batch row this lane loads A for

    v8f acc0 = {}, acc1 = {}, acc2 = {}, acc3 = {};

    // Weight row bases for the four gates (i, f, g, o) at column slice j0+ncl
    const half_t* w_i = wcat + (size_t)(0 * HH + j0 + ncl) * KCAT;
    const half_t* w_f = wcat + (size_t)(1 * HH + j0 + ncl) * KCAT;
    const half_t* w_g = wcat + (size_t)(2 * HH + j0 + ncl) * KCAT;
    const half_t* w_o = wcat + (size_t)(3 * HH + j0 + ncl) * KCAT;

    // ---- Part 1: x contribution, K = 0..127 (fp32 -> f16 in-register) ----
    const float* xrow = inputs + (size_t)mrow * TT * DD + (size_t)t * DD;
#pragma unroll
    for (int kk = 0; kk < DD; kk += 32) {
        const float* xp = xrow + kk + hi * 16;
        v8f xlo = *(const v8f*)(xp);
        v8f xhi2 = *(const v8f*)(xp + 8);
        v16h a;
#pragma unroll
        for (int i = 0; i < 8; ++i) { a[i] = (half_t)xlo[i]; a[8 + i] = (half_t)xhi2[i]; }
        const int kb = kk + hi * 16;
        v16h b0 = *(const v16h*)(w_i + kb);
        v16h b1 = *(const v16h*)(w_f + kb);
        v16h b2 = *(const v16h*)(w_g + kb);
        v16h b3 = *(const v16h*)(w_o + kb);
        acc0 = __builtin_amdgcn_wmma_f32_16x16x32_f16(false, a, false, b0, (short)0, acc0, false, false);
        acc1 = __builtin_amdgcn_wmma_f32_16x16x32_f16(false, a, false, b1, (short)0, acc1, false, false);
        acc2 = __builtin_amdgcn_wmma_f32_16x16x32_f16(false, a, false, b2, (short)0, acc2, false, false);
        acc3 = __builtin_amdgcn_wmma_f32_16x16x32_f16(false, a, false, b3, (short)0, acc3, false, false);
    }

    // ---- Part 2: h contribution, K = 0..511 (f16), weight cols offset by D ----
    for (int kk = 0; kk < HH; kk += 32) {
        const half_t* hp = hprev + (size_t)mrow * HH + kk + hi * 16;
        v8h alo = *(const v8h*)(hp);
        v8h ahi = *(const v8h*)(hp + 8);
        v16h a;
#pragma unroll
        for (int i = 0; i < 8; ++i) { a[i] = alo[i]; a[8 + i] = ahi[i]; }
        const int kb = DD + kk + hi * 16;
        v16h b0 = *(const v16h*)(w_i + kb);
        v16h b1 = *(const v16h*)(w_f + kb);
        v16h b2 = *(const v16h*)(w_g + kb);
        v16h b3 = *(const v16h*)(w_o + kb);
        acc0 = __builtin_amdgcn_wmma_f32_16x16x32_f16(false, a, false, b0, (short)0, acc0, false, false);
        acc1 = __builtin_amdgcn_wmma_f32_16x16x32_f16(false, a, false, b1, (short)0, acc1, false, false);
        acc2 = __builtin_amdgcn_wmma_f32_16x16x32_f16(false, a, false, b2, (short)0, acc2, false, false);
        acc3 = __builtin_amdgcn_wmma_f32_16x16x32_f16(false, a, false, b3, (short)0, acc3, false, false);
    }

    // ---- Fused gate nonlinearity + state update ----
    const int j = j0 + ncl;
    const float bi = biascat[0 * HH + j];
    const float bf = biascat[1 * HH + j];
    const float bg = biascat[2 * HH + j];
    const float bo = biascat[3 * HH + j];
#pragma unroll
    for (int r = 0; r < 8; ++r) {
        const int m = m0 + hi * 8 + r;
        const size_t idx = (size_t)m * HH + j;
        const float gi = sigf(acc0[r] + bi);
        const float gf = sigf(acc1[r] + bf);
        const float gg = tanhf(acc2[r] + bg);
        const float go = sigf(acc3[r] + bo);
        const float cn = gf * c[idx] + gi * gg;
        c[idx] = cn;
        hnext[idx] = (half_t)(go * tanhf(cn));
    }
}

// ---------------------------------------------------------------------------
// Prep / small kernels
// ---------------------------------------------------------------------------
__global__ void build_wcat_kernel(const float* __restrict__ W_ih,
                                  const float* __restrict__ W_hh,
                                  half_t* __restrict__ wcat, int n)
{
    for (int idx = blockIdx.x * blockDim.x + threadIdx.x; idx < n;
         idx += gridDim.x * blockDim.x) {
        const int row = idx / KCAT, col = idx % KCAT;
        const float v = (col < DD) ? W_ih[(size_t)row * DD + col]
                                   : W_hh[(size_t)row * HH + (col - DD)];
        wcat[idx] = (half_t)v;
    }
}

__global__ void vec_add_kernel(const float* __restrict__ a, const float* __restrict__ b,
                               float* __restrict__ out, int n)
{
    int idx = blockIdx.x * blockDim.x + threadIdx.x;
    if (idx < n) out[idx] = a[idx] + b[idx];
}

__global__ void conv_f16_kernel(const float* __restrict__ src, half_t* __restrict__ dst, int n)
{
    for (int idx = blockIdx.x * blockDim.x + threadIdx.x; idx < n;
         idx += gridDim.x * blockDim.x)
        dst[idx] = (half_t)src[idx];
}

__global__ void zero_init_kernel(float* __restrict__ c, half_t* __restrict__ h0)
{
    int idx = blockIdx.x * blockDim.x + threadIdx.x;      // B*H
    if (idx >= BB * HH) return;
    c[idx] = 0.0f;
    h0[idx] = (half_t)0.0f;
}

__global__ void z_init_kernel(const float* __restrict__ c, float* __restrict__ z,
                              half_t* __restrict__ zh, half_t* __restrict__ pred)
{
    int idx = blockIdx.x * blockDim.x + threadIdx.x;      // B*H
    if (idx >= BB * HH) return;
    const int b = idx >> 9, j = idx & 511;
    const float v = c[idx];
    z[idx] = v;
    zh[idx] = (half_t)v;
    pred[((size_t)b * NSTEPS) * HH + j] = (half_t)v;      // pred step 0 = z0
}

// ---------------------------------------------------------------------------
static void launch_gemm(hipStream_t s, const half_t* A, int lda, const half_t* W, int ldw,
                        const float* bias, float* Cf, half_t* Ch, int ldc,
                        int M, int N, int K, int act, int epi = EPI_PLAIN,
                        const float* zvec = nullptr, float alpha = 0.0f,
                        half_t* stage_h = nullptr,
                        const float* q1 = nullptr, const float* q2 = nullptr,
                        const float* q3 = nullptr, float dtd6 = 0.0f,
                        float* zout = nullptr, half_t* zhout = nullptr,
                        half_t* pred = nullptr, int sidx = 0)
{
    const int tiles = (M / 16) * (N / 64);
    const int blocks = (tiles + 3) / 4;
    wmma_gemm_kernel<<<blocks, 128, 0, s>>>(A, lda, W, ldw, bias, Cf, Ch, ldc, M, N, K,
                                            act, epi, zvec, alpha, stage_h,
                                            q1, q2, q3, dtd6, zout, zhout, pred, sidx);
}

extern "C" void kernel_launch(void* const* d_in, const int* in_sizes, int n_in,
                              void* d_out, int out_size, void* d_ws, size_t ws_size,
                              hipStream_t stream)
{
    (void)in_sizes; (void)n_in; (void)out_size; (void)ws_size;
    const float* inputs = (const float*)d_in[0];
    const float* W_ih   = (const float*)d_in[3];
    const float* W_hh   = (const float*)d_in[4];
    const float* b_ih   = (const float*)d_in[5];
    const float* b_hh   = (const float*)d_in[6];
    const float* ode_w1 = (const float*)d_in[7];
    const float* ode_b1 = (const float*)d_in[8];
    const float* ode_w2 = (const float*)d_in[9];
    const float* ode_b2 = (const float*)d_in[10];
    const float* ode_w3 = (const float*)d_in[11];
    const float* ode_b3 = (const float*)d_in[12];
    const float* dec_w1 = (const float*)d_in[13];
    const float* dec_b1 = (const float*)d_in[14];
    const float* dec_w2 = (const float*)d_in[15];
    const float* dec_b2 = (const float*)d_in[16];
    float* out = (float*)d_out;

    // ---- carve workspace (256B aligned sub-buffers) ----
    char* base = (char*)d_ws;
    size_t off = 0;
    auto carve = [&](size_t bytes) -> void* {
        void* p = base + off;
        off = (off + bytes + 255) & ~(size_t)255;
        return p;
    };
    half_t* wcat    = (half_t*)carve((size_t)G4 * KCAT * 2);       // [4H, D+H] f16
    float*  biascat = (float*) carve((size_t)G4 * 4);              // b_ih+b_hh
    half_t* hbuf0   = (half_t*)carve((size_t)BB * HH * 2);         // h ping
    half_t* hbuf1   = (half_t*)carve((size_t)BB * HH * 2);         // h pong
    float*  cst     = (float*) carve((size_t)BB * HH * 4);         // cell state
    float*  z       = (float*) carve((size_t)BB * HH * 4);
    half_t* zh      = (half_t*)carve((size_t)BB * HH * 2);
    half_t* ztmp    = (half_t*)carve((size_t)BB * HH * 2);
    float*  k1      = (float*) carve((size_t)BB * HH * 4);
    float*  k2      = (float*) carve((size_t)BB * HH * 4);
    float*  k3      = (float*) carve((size_t)BB * HH * 4);
    half_t* u1h     = (half_t*)carve((size_t)BB * NHH * 2);
    half_t* u2h     = (half_t*)carve((size_t)BB * NHH * 2);
    half_t* ow1h    = (half_t*)carve((size_t)NHH * HH * 2);
    half_t* ow2h    = (half_t*)carve((size_t)NHH * NHH * 2);
    half_t* ow3h    = (half_t*)carve((size_t)HH * NHH * 2);
    half_t* dw1h    = (half_t*)carve((size_t)NHH * HH * 2);
    half_t* dw2h    = (half_t*)carve((size_t)DD * NHH * 2);
    half_t* predzh  = (half_t*)carve((size_t)BB * NSTEPS * HH * 2);
    half_t* duh     = (half_t*)carve((size_t)BB * NSTEPS * NHH * 2);

    // ---- one-time prep (re-done every call: deterministic) ----
    build_wcat_kernel<<<1024, 256, 0, stream>>>(W_ih, W_hh, wcat, G4 * KCAT);
    vec_add_kernel<<<(G4 + 255) / 256, 256, 0, stream>>>(b_ih, b_hh, biascat, G4);
    conv_f16_kernel<<<1024, 256, 0, stream>>>(ode_w1, ow1h, NHH * HH);
    conv_f16_kernel<<<1024, 256, 0, stream>>>(ode_w2, ow2h, NHH * NHH);
    conv_f16_kernel<<<1024, 256, 0, stream>>>(ode_w3, ow3h, HH * NHH);
    conv_f16_kernel<<<1024, 256, 0, stream>>>(dec_w1, dw1h, NHH * HH);
    conv_f16_kernel<<<1024, 256, 0, stream>>>(dec_w2, dw2h, DD * NHH);
    const int ebh = (BB * HH + 255) / 256;
    zero_init_kernel<<<ebh, 256, 0, stream>>>(cst, hbuf0);

    // ---- LSTM encoder: one fused kernel per step (h ping-pongs) ----
    const int lstm_blocks = ((BB / 16) * (HH / 16) + 3) / 4;   // 256 waves -> 64 blocks
    for (int t = 0; t < TT; ++t) {
        const half_t* hp = (t & 1) ? hbuf1 : hbuf0;
        half_t*       hn = (t & 1) ? hbuf0 : hbuf1;
        lstm_gemm_kernel<<<lstm_blocks, 128, 0, stream>>>(inputs, t, hp, wcat,
                                                          biascat, cst, hn);
    }

    // ---- z0 = c_n ----
    z_init_kernel<<<ebh, 256, 0, stream>>>(cst, z, zh, predzh);

    // ---- latent ODE: fixed-step RK4, pointwise work fused into gemm3 ----
    const float dt = (5.0f / 60.0f) / (float)NSUB;
    auto mlp12 = [&](const half_t* zin) {
        launch_gemm(stream, zin, HH, ow1h, HH, ode_b1, nullptr, u1h, NHH,
                    BB, NHH, HH, ACT_ELU);
        launch_gemm(stream, u1h, NHH, ow2h, NHH, ode_b2, nullptr, u2h, NHH,
                    BB, NHH, NHH, ACT_ELU);
    };

    for (int step = 1; step < NSTEPS; ++step) {
        for (int sub = 0; sub < NSUB; ++sub) {
            // stage 1..3: k_i = f(.), ztmp = f16(z + alpha*k_i)
            mlp12(zh);
            launch_gemm(stream, u2h, NHH, ow3h, NHH, ode_b3, k1, nullptr, HH,
                        BB, HH, NHH, ACT_NONE, EPI_STAGE, z, 0.5f * dt, ztmp);
            mlp12(ztmp);
            launch_gemm(stream, u2h, NHH, ow3h, NHH, ode_b3, k2, nullptr, HH,
                        BB, HH, NHH, ACT_NONE, EPI_STAGE, z, 0.5f * dt, ztmp);
            mlp12(ztmp);
            launch_gemm(stream, u2h, NHH, ow3h, NHH, ode_b3, k3, nullptr, HH,
                        BB, HH, NHH, ACT_NONE, EPI_STAGE, z, dt, ztmp);
            // stage 4: fused RK4 combine (k4 never hits memory)
            mlp12(ztmp);
            half_t* pred = (sub == NSUB - 1) ? predzh : nullptr;
            launch_gemm(stream, u2h, NHH, ow3h, NHH, ode_b3, nullptr, nullptr, HH,
                        BB, HH, NHH, ACT_NONE, EPI_COMBINE, z, 0.0f, nullptr,
                        k1, k2, k3, dt / 6.0f, z, zh, pred, step);
        }
    }

    // ---- decoder: [B*5, H] -> ReLU [B*5, NH] -> [B*5, D] == d_out [B,5,D] ----
    launch_gemm(stream, predzh, HH, dw1h, HH, dec_b1, nullptr, duh, NHH,
                BB * NSTEPS, NHH, HH, ACT_RELU);
    launch_gemm(stream, duh, NHH, dw2h, NHH, dec_b2, out, nullptr, DD,
                BB * NSTEPS, DD, NHH, ACT_NONE);
}
